// ModulatedConv2d_52999896432961
// MI455X (gfx1250) — matrix operand order repaired and checked by reading, not compiled, not run on hardware
//
#include <hip/hip_runtime.h>
#include <hip/hip_bf16.h>
#include <stdint.h>

#define B_    16
#define CIN_  512
#define COUT_ 512
#define SD_   512
#define H_    64
#define W_    64
#define EPS_  1e-8f

typedef __attribute__((ext_vector_type(16))) __bf16 v16bf;
typedef __attribute__((ext_vector_type(8)))  __bf16 v8bf;
typedef __attribute__((ext_vector_type(8)))  float  v8f;
typedef __attribute__((ext_vector_type(4)))  int    v4i;

#if defined(__has_builtin)
#if __has_builtin(__builtin_amdgcn_global_load_async_to_lds_b128)
#define USE_ASYNC_LDS 1
#else
#define USE_ASYNC_LDS 0
#endif
#else
#define USE_ASYNC_LDS 0
#endif

#if USE_ASYNC_LDS
__device__ __forceinline__ void async_copy_b128(const unsigned short* g, unsigned short* l) {
    __builtin_amdgcn_global_load_async_to_lds_b128(
        (__attribute__((address_space(1))) v4i*)g,
        (__attribute__((address_space(3))) v4i*)l, 0, 0);
}
#endif

__device__ __forceinline__ unsigned short f2bf_rne(float f) {
    uint32_t u = __builtin_bit_cast(uint32_t, f);
    u += 0x7fffu + ((u >> 16) & 1u);
    return (unsigned short)(u >> 16);
}

// ---------------- Kernel 1: s[b,cin] = style @ (mod_weight * 1/sqrt(512)).T + bias
__global__ void style_mod_kernel(const float* __restrict__ style,
                                 const float* __restrict__ mw,
                                 const float* __restrict__ mb,
                                 float* __restrict__ s) {
    int idx = blockIdx.x * blockDim.x + threadIdx.x;       // b*CIN + cin
    if (idx >= B_ * CIN_) return;
    int b = idx / CIN_, cin = idx % CIN_;
    const float ms = 0.04419417382415922f;                  // 1/sqrt(512)
    const float* st = style + b * SD_;
    const float* wr = mw + (size_t)cin * SD_;
    float acc = 0.f;
#pragma unroll 8
    for (int j = 0; j < SD_; ++j) acc += st[j] * wr[j];
    s[idx] = acc * ms + mb[cin];
}

// ---------------- Kernel 2: bf16 weights reordered [cout][tap][cin] + wsq[cout][cin]=sum_kk w^2
__global__ void wprep_kernel(const float* __restrict__ w,
                             unsigned short* __restrict__ wbf,
                             float* __restrict__ wsq) {
    int idx = blockIdx.x * blockDim.x + threadIdx.x;       // cout*CIN + cin
    if (idx >= COUT_ * CIN_) return;
    int cout = idx / CIN_, cin = idx % CIN_;
    const float* wp = w + (size_t)idx * 9;
    float ss = 0.f;
#pragma unroll
    for (int t = 0; t < 9; ++t) {
        float v = wp[t];
        ss += v * v;
        wbf[((size_t)cout * 9 + t) * CIN_ + cin] = f2bf_rne(v);
    }
    wsq[idx] = ss;
}

// ---------------- Kernel 3: dscale[b,cout] = conv_scale * rsqrt(conv_scale^2 * sum wsq*s^2 + eps)
__global__ void demod_kernel(const float* __restrict__ s,
                             const float* __restrict__ wsq,
                             float* __restrict__ dscale) {
    int idx = blockIdx.x * blockDim.x + threadIdx.x;       // b*COUT + cout
    if (idx >= B_ * COUT_) return;
    int b = idx / COUT_, cout = idx % COUT_;
    const float cs = 0.014731391274719739f;                 // 1/sqrt(512*9)
    const float* sr = s + b * CIN_;
    const float* wr = wsq + (size_t)cout * CIN_;
    float acc = 0.f;
#pragma unroll 8
    for (int j = 0; j < CIN_; ++j) { float sv = sr[j]; acc += wr[j] * sv * sv; }
    dscale[idx] = cs * rsqrtf(cs * cs * acc + EPS_);
}

// ---------------- Kernel 4: xs[b,cin,h,w] = bf16(x * s[b,cin])  (hoists style scale out of conv)
__global__ void xscale_kernel(const float* __restrict__ x,
                              const float* __restrict__ s,
                              unsigned short* __restrict__ xs) {
    int idx = blockIdx.x * blockDim.x + threadIdx.x;        // 8 elems per thread
    size_t base = (size_t)idx * 8;
    float sc = s[base >> 12];                               // /(H*W)
    const float4* xp = (const float4*)(x + base);
    float4 v0 = xp[0], v1 = xp[1];
    uint4 o;
    o.x = (uint32_t)f2bf_rne(v0.x * sc) | ((uint32_t)f2bf_rne(v0.y * sc) << 16);
    o.y = (uint32_t)f2bf_rne(v0.z * sc) | ((uint32_t)f2bf_rne(v0.w * sc) << 16);
    o.z = (uint32_t)f2bf_rne(v1.x * sc) | ((uint32_t)f2bf_rne(v1.y * sc) << 16);
    o.w = (uint32_t)f2bf_rne(v1.z * sc) | ((uint32_t)f2bf_rne(v1.w * sc) << 16);
    *(uint4*)(xs + base) = o;
}

// ---------------- Kernel 5: implicit-GEMM conv, double-buffered LDS, v_wmma_f32_16x16x32_bf16
__launch_bounds__(256, 2)
__global__ void modconv_wmma_kernel(const unsigned short* __restrict__ xs,
                                    const unsigned short* __restrict__ wbf,
                                    const float* __restrict__ dscale,
                                    float* __restrict__ out) {
    // A: [buf][tap][128 cout x 32 k] stride 40; B: [buf][66 padded-w x 32 cin] stride 40
    __shared__ alignas(16) unsigned short As[2][3][128 * 40];
    __shared__ alignas(16) unsigned short Bs[2][66 * 40];

    const int tid   = threadIdx.x;
    const int lane  = tid & 31;
    const int wid   = tid >> 5;
    const int waveM = wid & 3;       // 4 waves x 32 rows -> 128 cout
    const int waveN = wid >> 2;      // 2 waves x 32 cols -> 64 w
    const int lrow  = lane & 15;
    const int lhalf = lane >> 4;

    const int cout0 = blockIdx.x * 128;
    const int h     = blockIdx.y;
    const int b     = blockIdx.z;

    // zero the w = -1 / w = 64 padding columns of both B buffers (never re-written)
    if (tid < 128) {
        int bufz = tid >> 6;
        int cz   = tid & 31;
        int pwz  = (tid & 32) ? 65 : 0;
        Bs[bufz][pwz * 40 + cz] = 0;
    }

    v8f acc[2][2] = {};

    const int ar   = tid >> 1;           // A loader row 0..127
    const int ac16 = (tid & 1) * 16;     // A loader col 0/16
    const int bc   = tid >> 3;           // B loader cin lane 0..31
    const int bw0  = (tid & 7) * 8;      // B loader w start (8 each)

    __syncthreads();

    // ki enumerates (dh, cin-chunk): 3 * 16 = 48 stages; software pipeline depth 1
    for (int ki = 0; ki <= 48; ++ki) {
        // ---------- stage tile ki into buffer ki&1
        if (ki < 48) {
            const int dh = ki >> 4;
            const int xh = h + dh - 1;
            if ((unsigned)xh < (unsigned)H_) {               // WG-uniform
                const int buf = ki & 1;
                const int kc  = (ki & 15) * 32;
                // A: 3 taps (dw) of 128x32 bf16 weights (pure copy -> async path)
                const unsigned short* gA =
                    wbf + ((size_t)(cout0 + ar) * 9 + dh * 3) * CIN_ + kc + ac16;
#pragma unroll
                for (int tap = 0; tap < 3; ++tap) {
                    unsigned short* lA = &As[buf][tap][ar * 40 + ac16];
#if USE_ASYNC_LDS
                    async_copy_b128(gA, lA);
                    async_copy_b128(gA + 8, lA + 8);
#else
                    uint4 v0 = *(const uint4*)gA;
                    uint4 v1 = *(const uint4*)(gA + 8);
                    *(uint4*)lA       = v0;
                    *(uint4*)(lA + 8) = v1;
#endif
                    gA += CIN_;
                }
                // B: transpose-scatter prescaled activations, Bs[w+1][c]
                {
                    const unsigned short* gB =
                        xs + (((size_t)(b * CIN_ + kc + bc)) * H_ + xh) * W_ + bw0;
                    uint4 v = *(const uint4*)gB;
                    unsigned short e[8];
                    e[0] = v.x & 0xffff; e[1] = v.x >> 16;
                    e[2] = v.y & 0xffff; e[3] = v.y >> 16;
                    e[4] = v.z & 0xffff; e[5] = v.z >> 16;
                    e[6] = v.w & 0xffff; e[7] = v.w >> 16;
#pragma unroll
                    for (int j = 0; j < 8; ++j)
                        Bs[buf][(bw0 + 1 + j) * 40 + bc] = e[j];
                }
            }
        }
        // ---------- compute tile ki-1 from buffer (ki-1)&1
        if (ki > 0) {
            const int kj = ki - 1;
            const int dh = kj >> 4;
            const int xh = h + dh - 1;
            if ((unsigned)xh < (unsigned)H_) {               // WG-uniform
                const int pbuf = kj & 1;
#pragma unroll
                for (int dw = 0; dw < 3; ++dw) {
                    v16bf afr[2];
#pragma unroll
                    for (int mi = 0; mi < 2; ++mi) {
                        const unsigned short* ap =
                            &As[pbuf][dw][(32 * waveM + 16 * mi + lrow) * 40 + 16 * lhalf];
                        v8bf lo = *(const v8bf*)ap;          // K 0-7 / 16-23
                        v8bf hi = *(const v8bf*)(ap + 8);    // K 8-15 / 24-31
                        v16bf a;
#pragma unroll
                        for (int e = 0; e < 8; ++e) { a[e] = lo[e]; a[e + 8] = hi[e]; }
                        afr[mi] = a;
                    }
#pragma unroll
                    for (int ni = 0; ni < 2; ++ni) {
                        // output w -> input pw = w + (dw-1) + 1 = w + dw
                        const int pw = 32 * waveN + 16 * ni + lrow + dw;
                        const v16bf bfr =
                            *(const v16bf*)&Bs[pbuf][pw * 40 + 16 * lhalf];
#pragma unroll
                        for (int mi = 0; mi < 2; ++mi)
                            acc[mi][ni] = __builtin_amdgcn_wmma_f32_16x16x32_bf16(
                                false, afr[mi], false, bfr,
                                (short)0, acc[mi][ni], false, false);
                    }
                }
            }
        }
#if USE_ASYNC_LDS
        asm volatile("s_wait_asynccnt 0x0" ::: "memory");
#endif
        __syncthreads();
    }

    // epilogue: scale by conv_scale*demod; C layout: VGPR v -> M = v + 8*lhalf, N = lrow
#pragma unroll
    for (int mi = 0; mi < 2; ++mi) {
#pragma unroll
        for (int v = 0; v < 8; ++v) {
            const int m    = 32 * waveM + 16 * mi + v + 8 * lhalf;
            const int cout = cout0 + m;
            const float sc = dscale[b * COUT_ + cout];
#pragma unroll
            for (int ni = 0; ni < 2; ++ni) {
                const int w = 32 * waveN + 16 * ni + lrow;
                out[(((size_t)b * COUT_ + cout) * H_ + h) * W_ + w] = acc[mi][ni][v] * sc;
            }
        }
    }
}

extern "C" void kernel_launch(void* const* d_in, const int* in_sizes, int n_in,
                              void* d_out, int out_size, void* d_ws, size_t ws_size,
                              hipStream_t stream) {
    (void)in_sizes; (void)n_in; (void)out_size; (void)ws_size;
    const float* x      = (const float*)d_in[0];
    const float* style  = (const float*)d_in[1];
    const float* weight = (const float*)d_in[2];
    const float* mw     = (const float*)d_in[3];
    const float* mb     = (const float*)d_in[4];
    float* out = (float*)d_out;

    char* ws = (char*)d_ws;
    float*          s_buf  = (float*)ws;                               // 16*512 f32   (32 KB)
    float*          dscale = (float*)(ws + 32768);                     // 16*512 f32   (32 KB)
    float*          wsq    = (float*)(ws + 65536);                     // 512*512 f32  (1 MB)
    unsigned short* wbf    = (unsigned short*)(ws + 65536 + 1048576);  // 512*9*512 bf16 (4.5 MB)
    unsigned short* xsb    = (unsigned short*)(ws + 65536 + 1048576 + 4718592); // 67 MB bf16

    style_mod_kernel<<<(B_ * CIN_ + 255) / 256, 256, 0, stream>>>(style, mw, mb, s_buf);
    wprep_kernel<<<(COUT_ * CIN_ + 255) / 256, 256, 0, stream>>>(weight, wbf, wsq);
    demod_kernel<<<(B_ * COUT_ + 255) / 256, 256, 0, stream>>>(s_buf, wsq, dscale);
    xscale_kernel<<<(B_ * CIN_ * H_ * W_ / 8 + 255) / 256, 256, 0, stream>>>(x, s_buf, xsb);

    dim3 grid(COUT_ / 128, H_, B_);
    modconv_wmma_kernel<<<grid, 256, 0, stream>>>(xsb, wbf, dscale, out);
}
